// SlidingATTN_12506944766460
// MI455X (gfx1250) — compile-verified, matrix operand order checked by reading, not run on hardware
//
#include <hip/hip_runtime.h>
#include <cstdint>
#include <cstddef>

// ---------------------------------------------------------------------------
// SlidingATTN for MI455X (gfx1250, wave32, WMMA).
// Two 8192x512x512 GEMMs on v_wmma_f32_16x16x32_f16 (f16 in, f32 acc).
// Weight matrices are [N][K] row-major, which is exactly the B-stage layout
// ([n][k] in LDS) -> both A and B tiles stage with contiguous b128 loads.
// Block tile 128x128, wave tile 32x64 -> 8 wmma per wave per k-step.
// ---------------------------------------------------------------------------

#define Bb   16
#define Ss   512
#define Cc   512
#define Hh   8
#define Ww   32
#define Dd   64
#define BS   (Bb * Ss)          // 8192
#define W1   (Ww + 1)           // 33
#define POSN (W1 * Hh)          // 264

typedef _Float16 f16;
typedef __attribute__((ext_vector_type(16))) _Float16 v16h;
typedef __attribute__((ext_vector_type(8)))  _Float16 v8h;
typedef __attribute__((ext_vector_type(8)))  float    v8f;

__device__ __forceinline__ float elu1(float z) {
    return z > 0.f ? z : (__expf(z) - 1.f);
}

// ---------------------------------------------------------------------------
// Conversions
// ---------------------------------------------------------------------------
__global__ void cvt_f32_f16(const float* __restrict__ in, f16* __restrict__ out, int n) {
    int i = blockIdx.x * blockDim.x + threadIdx.x;
    if (i < n) out[i] = (f16)in[i];
}

__global__ void zero_f32(float* __restrict__ p, int n) {
    int i = blockIdx.x * blockDim.x + threadIdx.x;
    if (i < n) p[i] = 0.f;
}

// ---------------------------------------------------------------------------
// WMMA GEMM: Y[M,N] = A[M,K](f16) * Bw[N,K](f16)^T + bias[N]  (f32 out)
// Block: 256 threads = 8 wave32s. Block tile 128(M) x 128(N).
// Wave tile 32(M) x 64(N): 2 A-frags x 4 B-frags = 8 v_wmma per k-step.
// ---------------------------------------------------------------------------
__global__ __launch_bounds__(256) void gemm_wmma_f16(
    const f16* __restrict__ A,   // [M][K] row-major
    const f16* __restrict__ Bw,  // [N][K] row-major (weight as-is)
    const float* __restrict__ bias, float* __restrict__ Y,
    int M, int N, int K)
{
    __shared__ __align__(16) f16 As[128][32];  // [m][k]
    __shared__ __align__(16) f16 Bs[128][32];  // [n][k]

    const int tid   = threadIdx.x;
    const int lane  = tid & 31;
    const int wave  = tid >> 5;
    const int m0    = blockIdx.x * 128;
    const int n0    = blockIdx.y * 128;
    const int msub  = (wave >> 1) * 32;   // 32-row group per wave
    const int ngrp  = (wave & 1) * 64;    // 64-col group per wave
    const int khalf = lane >> 4;          // ISA 16-bit frag: lane<16 -> K{0..7,16..23}
    const int l16   = lane & 15;

    v8f acc[2][4] = {};

    // staging coords: thread -> row tid>>1, 16 f16 at (tid&1)*16 (two b128s)
    const int srow  = tid >> 1;
    const int shalf = (tid & 1) * 16;

    for (int k0 = 0; k0 < K; k0 += 32) {
        {
            const v8h* src = (const v8h*)(A + (size_t)(m0 + srow) * K + k0 + shalf);
            *(v8h*)&As[srow][shalf]     = src[0];
            *(v8h*)&As[srow][shalf + 8] = src[1];
        }
        {
            const v8h* src = (const v8h*)(Bw + (size_t)(n0 + srow) * K + k0 + shalf);
            *(v8h*)&Bs[srow][shalf]     = src[0];
            *(v8h*)&Bs[srow][shalf + 8] = src[1];
        }
        __syncthreads();

        // A fragments (two 16x32 tiles), per ISA 7.12.2 layout
        v16h af[2];
        #pragma unroll
        for (int mi = 0; mi < 2; ++mi) {
            v8h lo = *(const v8h*)&As[msub + mi * 16 + l16][khalf * 8];
            v8h hi = *(const v8h*)&As[msub + mi * 16 + l16][16 + khalf * 8];
            af[mi] = __builtin_shufflevector(lo, hi,
                0,1,2,3,4,5,6,7,8,9,10,11,12,13,14,15);
        }
        #pragma unroll
        for (int ti = 0; ti < 4; ++ti) {
            int ncol = ngrp + ti * 16 + l16;
            v8h lo = *(const v8h*)&Bs[ncol][khalf * 8];
            v8h hi = *(const v8h*)&Bs[ncol][16 + khalf * 8];
            v16h bf = __builtin_shufflevector(lo, hi,
                0,1,2,3,4,5,6,7,8,9,10,11,12,13,14,15);
            acc[0][ti] = __builtin_amdgcn_wmma_f32_16x16x32_f16(
                false, af[0], false, bf, (short)0, acc[0][ti], false, false);
            acc[1][ti] = __builtin_amdgcn_wmma_f32_16x16x32_f16(
                false, af[1], false, bf, (short)0, acc[1][ti], false, false);
        }
        __syncthreads();
    }

    // epilogue: C/D layout: VGPR r -> M = base + (lane>>4)*8 + r, N = l16
    #pragma unroll
    for (int ti = 0; ti < 4; ++ti) {
        int n  = n0 + ngrp + ti * 16 + l16;
        float bv = bias ? bias[n] : 0.f;
        #pragma unroll
        for (int mi = 0; mi < 2; ++mi) {
            #pragma unroll
            for (int r = 0; r < 8; ++r) {
                int m = m0 + msub + mi * 16 + khalf * 8 + r;
                Y[(size_t)m * N + n] = acc[mi][ti][r] + bv;
            }
        }
    }
}

// ---------------------------------------------------------------------------
// attn_exp[b,s,h] = exp(x[b,s,:]·attn_w[h,:] + attn_b[h]) * (mask? 0 : 1)
// plus atomic column-sum into denom[b,h].  One block per (b,s); wave = head.
// ---------------------------------------------------------------------------
__global__ __launch_bounds__(256) void attn_kernel(
    const float* __restrict__ x, const unsigned char* __restrict__ mask,
    const float* __restrict__ aw, const float* __restrict__ ab,
    float* __restrict__ attn_exp, float* __restrict__ denom)
{
    int row  = blockIdx.x;             // b*S + s
    int lane = threadIdx.x & 31;
    int h    = threadIdx.x >> 5;       // 8 waves = 8 heads
    const float* xr = x  + (size_t)row * Cc;
    const float* wr = aw + (size_t)h   * Cc;
    float s = 0.f;
    #pragma unroll
    for (int i = 0; i < 16; ++i) {
        int c = lane + 32 * i;
        s += xr[c] * wr[c];
    }
    #pragma unroll
    for (int o = 16; o > 0; o >>= 1) s += __shfl_xor(s, o, 32);
    if (lane == 0) {
        float nm  = mask[row] ? 0.f : 1.f;
        float val = __expf(s + ab[h]) * nm;
        attn_exp[(size_t)row * Hh + h] = val;
        atomicAdd(&denom[(row >> 9) * Hh + h], val);
    }
}

// ---------------------------------------------------------------------------
// g_pool[b,h,d] = (sum_s attn_exp[b,s,h] * v[b,s,h,d]) / (denom[b,h] + 1e-5)
// ---------------------------------------------------------------------------
__global__ void gpool_kernel(
    const float* __restrict__ attn_exp, const float* __restrict__ v,
    const float* __restrict__ denom, float* __restrict__ gpool)
{
    int bh = blockIdx.x;               // b*H + h
    int b  = bh >> 3, h = bh & 7;
    int d  = threadIdx.x;              // 0..63
    float acc = 0.f;
    for (int s = 0; s < Ss; ++s) {
        size_t row = (size_t)b * Ss + s;
        acc += attn_exp[row * Hh + h] * v[row * Cc + h * Dd + d];
    }
    gpool[(size_t)bh * Dd + d] = acc / (denom[bh] + 1e-5f);
}

// ---------------------------------------------------------------------------
// pos-MLP per batch: h1 = elu(LN(gp·w1^T + b1)); pos = exp(h1·w2^T + b2)
// ---------------------------------------------------------------------------
__global__ void pos_kernel(
    const float* __restrict__ gpool,
    const float* __restrict__ w1, const float* __restrict__ b1,
    const float* __restrict__ g1, const float* __restrict__ be1,
    const float* __restrict__ w2, const float* __restrict__ b2,
    float* __restrict__ pos)
{
    __shared__ float gp[Cc], part[64], h1v[Hh];
    __shared__ float mu_s, rs_s;
    int b = blockIdx.x, t = threadIdx.x;
    for (int i = t; i < Cc; i += 64) gp[i] = gpool[(size_t)b * Cc + i];
    __syncthreads();
    int h = t >> 3, pr = t & 7;
    float s = 0.f;
    for (int j = 0; j < 64; ++j) s += gp[pr * 64 + j] * w1[(size_t)h * Cc + pr * 64 + j];
    part[t] = s;
    __syncthreads();
    if (t < Hh) {
        float acc = b1[t];
        for (int j = 0; j < 8; ++j) acc += part[t * 8 + j];
        h1v[t] = acc;
    }
    __syncthreads();
    if (t == 0) {
        float m = 0.f;
        for (int i = 0; i < Hh; ++i) m += h1v[i];
        m *= (1.f / Hh);
        float vv = 0.f;
        for (int i = 0; i < Hh; ++i) { float dd = h1v[i] - m; vv += dd * dd; }
        vv *= (1.f / Hh);
        mu_s = m; rs_s = rsqrtf(vv + 1e-5f);
    }
    __syncthreads();
    if (t < Hh) h1v[t] = elu1((h1v[t] - mu_s) * rs_s * g1[t] + be1[t]);
    __syncthreads();
    for (int j = t; j < POSN; j += 64) {
        float acc = b2[j];
        #pragma unroll
        for (int k = 0; k < Hh; ++k) acc += h1v[k] * w2[j * Hh + k];
        pos[(size_t)b * POSN + j] = __expf(acc);
    }
}

// ---------------------------------------------------------------------------
// Sliding window combine. Faithful to the reference [:W] slice quirk:
//   aw_cat[0] = pos[b,0,h]; aw_cat[j] = attn_exp[b, idx[s,j-1], h]*pos[b,j,h]
//   den = sum_{j=0..W} aw_cat + 1e-5
//   out[c] = (aw_cat[0]/den)*g_pool[b,c] + sum_{j=0..W-1} (aw_cat[j]/den)*v[b, idx[s,j], c]
// ---------------------------------------------------------------------------
__global__ __launch_bounds__(256) void sliding_kernel(
    const float* __restrict__ attn_exp, const float* __restrict__ v,
    const float* __restrict__ pos, const float* __restrict__ gpool,
    const int* __restrict__ indices, float* __restrict__ out)
{
    __shared__ float aw[W1 * Hh];
    __shared__ float inv[Hh];
    __shared__ int   idxs[Ww];
    int bs = blockIdx.x;
    int b  = bs >> 9;
    int s  = bs & (Ss - 1);
    int t  = threadIdx.x;
    if (t < Ww) idxs[t] = indices[s * Ww + t];
    __syncthreads();
    if (t < POSN) {
        int j = t >> 3, h = t & 7;
        float val;
        if (j == 0) {
            val = pos[((size_t)b * W1 + 0) * Hh + h];
        } else {
            int r = idxs[j - 1];
            float ae = (r < Ss) ? attn_exp[(size_t)(b * Ss + r) * Hh + h] : 0.f;
            val = ae * pos[((size_t)b * W1 + j) * Hh + h];
        }
        aw[j * Hh + h] = val;
    }
    __syncthreads();
    if (t < Hh) {
        float d = 1e-5f;
        for (int j = 0; j < W1; ++j) d += aw[j * Hh + t];
        inv[t] = 1.f / d;
    }
    __syncthreads();
    size_t base = (size_t)bs * Cc;
    #pragma unroll
    for (int e = 0; e < 2; ++e) {
        int c = t + e * 256;
        int h = c >> 6;
        float iv  = inv[h];
        float acc = aw[h] * iv * gpool[(size_t)b * Cc + c];   // aw[0*H+h] with g_pool
        for (int j = 0; j < Ww; ++j) {
            int r = idxs[j];
            if (r < Ss) {
                const float* vr = v + (size_t)(b * Ss + r) * Cc;
                if (j + 1 < Ww) {
                    int r2 = idxs[j + 1];
                    if (r2 < Ss)
                        __builtin_prefetch(v + (size_t)(b * Ss + r2) * Cc + c, 0, 0);
                }
                acc += aw[j * Hh + h] * iv * vr[c];
            }
        }
        out[base + c] = acc;
    }
}

// ---------------------------------------------------------------------------
// y = elu(LN(in_row)) [+ residual_row]; one block per row of 512.
// ---------------------------------------------------------------------------
__global__ __launch_bounds__(256) void ln_elu_kernel(
    const float* __restrict__ in, const float* __restrict__ g,
    const float* __restrict__ be, const float* __restrict__ residual,
    float* __restrict__ out)
{
    __shared__ float r1[256], r2[256];
    int row = blockIdx.x, t = threadIdx.x;
    size_t base = (size_t)row * Cc;
    float a0 = in[base + t], a1 = in[base + t + 256];
    r1[t] = a0 + a1;
    r2[t] = a0 * a0 + a1 * a1;
    __syncthreads();
    for (int o = 128; o > 0; o >>= 1) {
        if (t < o) { r1[t] += r1[t + o]; r2[t] += r2[t + o]; }
        __syncthreads();
    }
    float mu  = r1[0] * (1.f / Cc);
    float var = r2[0] * (1.f / Cc) - mu * mu;
    float rs  = rsqrtf(var + 1e-5f);
    float y0 = elu1((a0 - mu) * rs * g[t]       + be[t]);
    float y1 = elu1((a1 - mu) * rs * g[t + 256] + be[t + 256]);
    if (residual) {
        y0 += residual[base + t];
        y1 += residual[base + t + 256];
    }
    out[base + t]       = y0;
    out[base + t + 256] = y1;
}

// ---------------------------------------------------------------------------
// Workspace layout (bytes, 256-aligned)
// ---------------------------------------------------------------------------
static constexpr size_t alignUp(size_t x) { return (x + 255) & ~size_t(255); }
static constexpr size_t OFF_X16   = 0;
static constexpr size_t SZ_X16    = alignUp((size_t)BS * Cc * 2);
static constexpr size_t OFF_VW16  = OFF_X16 + SZ_X16;
static constexpr size_t SZ_W16    = alignUp((size_t)Cc * Cc * 2);
static constexpr size_t OFF_OW16  = OFF_VW16 + SZ_W16;
static constexpr size_t OFF_V     = OFF_OW16 + SZ_W16;
static constexpr size_t SZ_F32    = alignUp((size_t)BS * Cc * 4);
static constexpr size_t OFF_AE    = OFF_V + SZ_F32;
static constexpr size_t SZ_AE     = alignUp((size_t)BS * Hh * 4);
static constexpr size_t OFF_DEN   = OFF_AE + SZ_AE;
static constexpr size_t SZ_DEN    = alignUp((size_t)Bb * Hh * 4);
static constexpr size_t OFF_GP    = OFF_DEN + SZ_DEN;
static constexpr size_t SZ_GP     = alignUp((size_t)Bb * Cc * 4);
static constexpr size_t OFF_POS   = OFF_GP + SZ_GP;
static constexpr size_t SZ_POS    = alignUp((size_t)Bb * POSN * 4);
static constexpr size_t OFF_OA    = OFF_POS + SZ_POS;  // sliding out; later reused as GEMM2 out
static constexpr size_t OFF_OA16  = OFF_OA + SZ_F32;

extern "C" void kernel_launch(void* const* d_in, const int* in_sizes, int n_in,
                              void* d_out, int out_size, void* d_ws, size_t ws_size,
                              hipStream_t stream) {
    (void)in_sizes; (void)n_in; (void)out_size; (void)ws_size;
    const float*         x       = (const float*)d_in[0];
    const unsigned char* mask    = (const unsigned char*)d_in[1];
    const float*         attn_w  = (const float*)d_in[2];
    const float*         attn_b  = (const float*)d_in[3];
    const float*         pos_w1  = (const float*)d_in[4];
    const float*         pos_b1  = (const float*)d_in[5];
    const float*         pos_g1  = (const float*)d_in[6];
    const float*         pos_be1 = (const float*)d_in[7];
    const float*         pos_w2  = (const float*)d_in[8];
    const float*         pos_b2  = (const float*)d_in[9];
    const float*         v_w     = (const float*)d_in[10];
    const float*         v_b     = (const float*)d_in[11];
    const float*         v_g     = (const float*)d_in[12];
    const float*         v_be    = (const float*)d_in[13];
    const float*         out_w   = (const float*)d_in[14];
    const float*         out_b   = (const float*)d_in[15];
    const float*         out_g   = (const float*)d_in[16];
    const float*         out_be  = (const float*)d_in[17];
    const int*           indices = (const int*)d_in[18];

    char* ws = (char*)d_ws;
    f16*   x16   = (f16*)(ws + OFF_X16);
    f16*   vw16  = (f16*)(ws + OFF_VW16);
    f16*   ow16  = (f16*)(ws + OFF_OW16);
    float* buf_v = (float*)(ws + OFF_V);
    float* ae    = (float*)(ws + OFF_AE);
    float* den   = (float*)(ws + OFF_DEN);
    float* gp    = (float*)(ws + OFF_GP);
    float* posb  = (float*)(ws + OFF_POS);
    float* oa    = (float*)(ws + OFF_OA);     // sliding output / out-proj GEMM output
    f16*   oa16  = (f16*)(ws + OFF_OA16);

    const int NXC = BS * Cc;
    const int NWW = Cc * Cc;
    // 1) precision conversions (weights stay [N][K] = B-stage layout) + denom zero
    cvt_f32_f16<<<(NXC + 255) / 256, 256, 0, stream>>>(x, x16, NXC);
    cvt_f32_f16<<<(NWW + 255) / 256, 256, 0, stream>>>(v_w, vw16, NWW);
    cvt_f32_f16<<<(NWW + 255) / 256, 256, 0, stream>>>(out_w, ow16, NWW);
    zero_f32<<<1, 128, 0, stream>>>(den, Bb * Hh);

    // 2) v projection GEMM (WMMA) + LN/ELU in-place
    dim3 gdim(BS / 128, Cc / 128);
    gemm_wmma_f16<<<gdim, 256, 0, stream>>>(x16, vw16, v_b, buf_v, BS, Cc, Cc);
    ln_elu_kernel<<<BS, 256, 0, stream>>>(buf_v, v_g, v_be, nullptr, buf_v);

    // 3) attention logits / pooling / pos-MLP
    attn_kernel<<<BS, 256, 0, stream>>>(x, mask, attn_w, attn_b, ae, den);
    gpool_kernel<<<Bb * Hh, 64, 0, stream>>>(ae, buf_v, den, gp);
    pos_kernel<<<Bb, 64, 0, stream>>>(gp, pos_w1, pos_b1, pos_g1, pos_be1,
                                      pos_w2, pos_b2, posb);

    // 4) sliding-window combine
    sliding_kernel<<<BS, 256, 0, stream>>>(ae, buf_v, posb, gp, indices, oa);

    // 5) output projection GEMM (WMMA) + LN/ELU + residual -> d_out
    cvt_f32_f16<<<(NXC + 255) / 256, 256, 0, stream>>>(oa, oa16, NXC);
    gemm_wmma_f16<<<gdim, 256, 0, stream>>>(oa16, ow16, out_b, oa, BS, Cc, Cc);
    ln_elu_kernel<<<BS, 256, 0, stream>>>(oa, out_g, out_be, x, (float*)d_out);
}
